// GraphMSG_55198919688856
// MI455X (gfx1250) — matrix operand order repaired and checked by reading, not compiled, not run on hardware
//
#include <hip/hip_runtime.h>
#include <hip/hip_bf16.h>

// ---------------------------------------------------------------------------
// AIFS-style GNN forward for MI455X (gfx1250, wave32, WMMA).
// All GEMMs run on v_wmma_f32_16x16x32_bf16 with f32 accumulation.
// Weights are repacked once per call into the native WMMA B-fragment layout
// so the GEMM inner loop is ds_load_b128 + global_load_b128 + v_wmma only.
// segment_sum is implemented with global_atomic_add_f32 scatter.
// ---------------------------------------------------------------------------

typedef __attribute__((ext_vector_type(16))) __bf16 v16bf;
typedef __attribute__((ext_vector_type(8)))  float  v8f;
typedef __attribute__((ext_vector_type(4)))  float  f4v;

#define HID     128
#define F_IN    100
#define MSTEP   2
#define N_ERA   40320
#define N_HID   10242
#define E_E2H   200000
#define E_H2E   120000
#define E_H2H   61440

// gather modes
enum { G_ERA = 0, G_NODEF = 1, G_EDGEF = 2, G_CAT3 = 3, G_CAT2 = 4, G_PLAIN = 5 };
// output modes
enum { O_F32 = 0, O_BF16 = 1, O_SCATTER = 2, O_RESID = 3 };

struct MlpArgs {
    const __bf16* w1p;      // packed (dinPad x 128) WMMA-B layout
    const __bf16* w2p;      // packed (128 x 128)   WMMA-B layout
    const float *b1, *b2, *g, *beta;
    const float *srcA, *srcB, *srcC;  // f32 gather sources
    const __bf16* srcE;               // bf16 gather source (edge embeddings)
    const int *sidx, *didx;           // edge index arrays
    float*  outF;
    __bf16* outB;
    const float* resBase;             // residual base (may alias outF)
    int rows;
    int dinPad;                       // multiple of 32
};

// ---------------------------------------------------------------------------
// Weight packer: f32 (din x dout) row-major -> bf16 WMMA B fragments.
// Layout: [kb][nb][lane 0..31][j 0..15], element = W[kb*32 + (lane>>4)*8 + (j>>3)*16 + (j&7),
// nb*16 + (lane&15)], zero-padded. Lane data is 32B contiguous.
// ---------------------------------------------------------------------------
__global__ __launch_bounds__(256) void pack_w_kernel(const float* __restrict__ W,
                                                     __bf16* __restrict__ out,
                                                     int din, int dinPad,
                                                     int dout, int doutPad) {
    const int total = dinPad * doutPad;
    for (int t = blockIdx.x * 256 + threadIdx.x; t < total; t += gridDim.x * 256) {
        const int j    = t & 15;
        const int lane = (t >> 4) & 31;
        const int tb   = t >> 9;                 // kb*(doutPad/16)+nb
        const int nbs  = doutPad >> 4;
        const int nb   = tb % nbs;
        const int kb   = tb / nbs;
        const int k    = (kb << 5) + ((lane >> 4) << 3) + ((j >> 3) << 4) + (j & 7);
        const int n    = (nb << 4) + (lane & 15);
        float v = (k < din && n < dout) ? W[(size_t)k * dout + n] : 0.f;
        out[t] = (__bf16)v;
    }
}

__global__ __launch_bounds__(256) void zero_kernel(float* p, int n) {
    int i = blockIdx.x * 256 + threadIdx.x;
    if (i < n) p[i] = 0.f;
}

// ---------------------------------------------------------------------------
// Fused 2-layer MLP (din -> 128 -> 128) + SiLU + LayerNorm, one 16-row tile
// per block, 8 waves cover the 128 output channels (16x16 WMMA tiles each).
// ---------------------------------------------------------------------------
template <int GM, int OM>
__global__ __launch_bounds__(256) void mlp2_kernel(MlpArgs a) {
    __shared__ __align__(16) __bf16 A_lds[16 * 384];
    __shared__ __align__(16) __bf16 H_lds[16 * HID];
    __shared__ __align__(16) float  O_lds[16 * HID];
    __shared__ float red1[256], red2[256], stat[32];

    const int t    = threadIdx.x;
    const int lane = t & 31;
    const int wave = t >> 5;                 // output N-tile (0..7)
    const int row0 = blockIdx.x << 4;
    const int dinPad = a.dinPad;

    // ---- gather row features into LDS as bf16 (4 features / thread-iter) ----
    const int cpr = dinPad >> 2;
    for (int idx = t; idx < 16 * cpr; idx += 256) {
        const int r    = idx / cpr;
        const int j4   = (idx - r * cpr) << 2;
        const int grow = row0 + r;
        float f0 = 0.f, f1 = 0.f, f2 = 0.f, f3 = 0.f;
        if (grow < a.rows) {
            if constexpr (GM == G_ERA) {
                if (j4 < MSTEP * F_IN) {           // x 'b m n f -> n (m f)'
                    const int m = j4 / F_IN, c = j4 - m * F_IN;
                    f4v v = *(const f4v*)(a.srcA + ((size_t)m * a.rows + grow) * F_IN + c);
                    f0 = v.x; f1 = v.y; f2 = v.z; f3 = v.w;
                } else if (j4 < 204) {             // era_latlons (4)
                    f4v v = *(const f4v*)(a.srcB + (size_t)grow * 4);
                    f0 = v.x; f1 = v.y; f2 = v.z; f3 = v.w;
                } else if (j4 < 212) {             // era_trainable (8)
                    f4v v = *(const f4v*)(a.srcC + (size_t)grow * 8 + (j4 - 204));
                    f0 = v.x; f1 = v.y; f2 = v.z; f3 = v.w;
                }
            } else if constexpr (GM == G_NODEF) {  // latlons(4) + trainable(8)
                if (j4 < 4) {
                    f4v v = *(const f4v*)(a.srcA + (size_t)grow * 4);
                    f0 = v.x; f1 = v.y; f2 = v.z; f3 = v.w;
                } else if (j4 < 12) {
                    f4v v = *(const f4v*)(a.srcB + (size_t)grow * 8 + (j4 - 4));
                    f0 = v.x; f1 = v.y; f2 = v.z; f3 = v.w;
                }
            } else if constexpr (GM == G_EDGEF) {  // attr(3) + trainable(8)
                float f[4];
                #pragma unroll
                for (int i = 0; i < 4; ++i) {
                    const int j = j4 + i;
                    f[i] = (j < 3) ? a.srcA[(size_t)grow * 3 + j]
                         : (j < 11) ? a.srcB[(size_t)grow * 8 + (j - 3)] : 0.f;
                }
                f0 = f[0]; f1 = f[1]; f2 = f[2]; f3 = f[3];
            } else if constexpr (GM == G_CAT3) {   // [xs[src], xd[dst], e]
                if (j4 < HID) {
                    const int s = a.sidx[grow];
                    f4v v = *(const f4v*)(a.srcA + (size_t)s * HID + j4);
                    f0 = v.x; f1 = v.y; f2 = v.z; f3 = v.w;
                } else if (j4 < 2 * HID) {
                    const int d = a.didx[grow];
                    f4v v = *(const f4v*)(a.srcB + (size_t)d * HID + (j4 - HID));
                    f0 = v.x; f1 = v.y; f2 = v.z; f3 = v.w;
                } else {
                    union { uint2 u; __bf16 h[4]; } e4;
                    e4.u = *(const uint2*)(a.srcE + (size_t)grow * HID + (j4 - 2 * HID));
                    f0 = (float)e4.h[0]; f1 = (float)e4.h[1];
                    f2 = (float)e4.h[2]; f3 = (float)e4.h[3];
                }
            } else if constexpr (GM == G_CAT2) {   // [xd, agg]
                const float* p = (j4 < HID) ? (a.srcA + (size_t)grow * HID + j4)
                                            : (a.srcB + (size_t)grow * HID + (j4 - HID));
                f4v v = *(const f4v*)p;
                f0 = v.x; f1 = v.y; f2 = v.z; f3 = v.w;
            } else {                                // G_PLAIN
                f4v v = *(const f4v*)(a.srcA + (size_t)grow * HID + j4);
                f0 = v.x; f1 = v.y; f2 = v.z; f3 = v.w;
            }
        }
        union { __bf16 h[4]; uint2 u; } pk;
        pk.h[0] = (__bf16)f0; pk.h[1] = (__bf16)f1;
        pk.h[2] = (__bf16)f2; pk.h[3] = (__bf16)f3;
        *(uint2*)&A_lds[r * dinPad + j4] = pk.u;
    }
    __syncthreads();

    // ---- layer 1: (16 x dinPad) @ w1 ----
    const int arow = lane & 15;
    const int koff = (lane >> 4) << 3;       // ISA 16-bit A layout: K run offset
    v8f acc = {0.f, 0.f, 0.f, 0.f, 0.f, 0.f, 0.f, 0.f};
    for (int kb = 0; kb < (dinPad >> 5); ++kb) {
        union { v16bf v; uint4 u[2]; } av, bv;
        const __bf16* ap = &A_lds[arow * dinPad + (kb << 5) + koff];
        av.u[0] = *(const uint4*)ap;          // K0+koff .. +7
        av.u[1] = *(const uint4*)(ap + 16);   // K0+16+koff .. +7
        const __bf16* bp = a.w1p + ((((size_t)kb * 8 + wave) * 32 + lane) << 4);
        bv.u[0] = *(const uint4*)bp;
        bv.u[1] = *(const uint4*)(bp + 8);
        __builtin_prefetch(bp + 4096, 0, 1);  // next K-step of this B panel
        acc = __builtin_amdgcn_wmma_f32_16x16x32_bf16(false, av.v, false, bv.v,
                                                      (short)0, acc, false, false);
    }
    {   // bias + SiLU -> H tile (bf16) in LDS
        const int n  = (wave << 4) + (lane & 15);
        const int mb = (lane >> 4) << 3;
        const float b = a.b1[n];
        #pragma unroll
        for (int i = 0; i < 8; ++i) {
            float v = acc[i] + b;
            v = v * (1.f / (1.f + __expf(-v)));
            H_lds[(mb + i) * HID + n] = (__bf16)v;
        }
    }
    __syncthreads();

    // ---- layer 2: (16 x 128) @ w2 ----
    v8f acc2 = {0.f, 0.f, 0.f, 0.f, 0.f, 0.f, 0.f, 0.f};
    #pragma unroll
    for (int kb = 0; kb < 4; ++kb) {
        union { v16bf v; uint4 u[2]; } av, bv;
        const __bf16* ap = &H_lds[arow * HID + (kb << 5) + koff];
        av.u[0] = *(const uint4*)ap;
        av.u[1] = *(const uint4*)(ap + 16);
        const __bf16* bp = a.w2p + ((((size_t)kb * 8 + wave) * 32 + lane) << 4);
        bv.u[0] = *(const uint4*)bp;
        bv.u[1] = *(const uint4*)(bp + 8);
        acc2 = __builtin_amdgcn_wmma_f32_16x16x32_bf16(false, av.v, false, bv.v,
                                                       (short)0, acc2, false, false);
    }
    {   // + b2 -> f32 tile in LDS for cross-wave LayerNorm
        const int n  = (wave << 4) + (lane & 15);
        const int mb = (lane >> 4) << 3;
        const float b = a.b2[n];
        #pragma unroll
        for (int i = 0; i < 8; ++i) O_lds[(mb + i) * HID + n] = acc2[i] + b;
    }
    __syncthreads();

    // ---- LayerNorm over 128 channels (16 threads per row) ----
    {
        const int r = t >> 4, seg = t & 15;
        const float* p = &O_lds[r * HID + (seg << 3)];
        float s = 0.f, q = 0.f;
        #pragma unroll
        for (int j = 0; j < 8; ++j) { float v = p[j]; s += v; q += v * v; }
        red1[t] = s; red2[t] = q;
    }
    __syncthreads();
    if (t < 16) {
        float s = 0.f, q = 0.f;
        #pragma unroll
        for (int j = 0; j < 16; ++j) { s += red1[(t << 4) + j]; q += red2[(t << 4) + j]; }
        const float mu = s * (1.f / HID);
        const float var = q * (1.f / HID) - mu * mu;
        stat[t * 2] = mu;
        stat[t * 2 + 1] = rsqrtf(var + 1e-5f);
    }
    __syncthreads();
    {
        const int r = t >> 4, seg = t & 15, grow = row0 + r;
        if (grow < a.rows) {
            const float mu = stat[r * 2], is = stat[r * 2 + 1];
            int dbase = 0;
            if constexpr (OM == O_SCATTER) dbase = a.didx[grow];
            #pragma unroll
            for (int j = 0; j < 8; ++j) {
                const int c = (seg << 3) + j;
                const float v = (O_lds[r * HID + c] - mu) * is * a.g[c] + a.beta[c];
                if constexpr (OM == O_F32)
                    a.outF[(size_t)grow * HID + c] = v;
                else if constexpr (OM == O_BF16)
                    a.outB[(size_t)grow * HID + c] = (__bf16)v;
                else if constexpr (OM == O_SCATTER)
                    atomicAdd(&a.outF[(size_t)dbase * HID + c], v);   // segment_sum
                else
                    a.outF[(size_t)grow * HID + c] =
                        a.resBase[(size_t)grow * HID + c] + v;        // residual
            }
        }
    }
}

// ---------------------------------------------------------------------------
// Final decoder head: out = XO @ out_w + out_b + x[:, -1]   (128 -> 100)
// ---------------------------------------------------------------------------
__global__ __launch_bounds__(256) void out_gemm_kernel(const float* __restrict__ XO,
                                                       const __bf16* __restrict__ wp,
                                                       const float* __restrict__ ob,
                                                       const float* __restrict__ x,
                                                       float* __restrict__ out,
                                                       int rows) {
    __shared__ __align__(16) __bf16 A_lds[16 * HID];
    const int t = threadIdx.x, lane = t & 31, wave = t >> 5;
    const int row0 = blockIdx.x << 4;

    for (int idx = t; idx < 16 * (HID >> 2); idx += 256) {
        const int r = idx >> 5, j4 = (idx & 31) << 2;
        const int grow = row0 + r;
        float f0 = 0.f, f1 = 0.f, f2 = 0.f, f3 = 0.f;
        if (grow < rows) {
            f4v v = *(const f4v*)(XO + (size_t)grow * HID + j4);
            f0 = v.x; f1 = v.y; f2 = v.z; f3 = v.w;
        }
        union { __bf16 h[4]; uint2 u; } pk;
        pk.h[0] = (__bf16)f0; pk.h[1] = (__bf16)f1;
        pk.h[2] = (__bf16)f2; pk.h[3] = (__bf16)f3;
        *(uint2*)&A_lds[r * HID + j4] = pk.u;
    }
    __syncthreads();

    const int arow = lane & 15;
    const int koff = (lane >> 4) << 3;
    v8f acc = {0.f, 0.f, 0.f, 0.f, 0.f, 0.f, 0.f, 0.f};
    #pragma unroll
    for (int kb = 0; kb < 4; ++kb) {
        union { v16bf v; uint4 u[2]; } av, bv;
        const __bf16* ap = &A_lds[arow * HID + (kb << 5) + koff];
        av.u[0] = *(const uint4*)ap;
        av.u[1] = *(const uint4*)(ap + 16);
        const __bf16* bp = wp + ((((size_t)kb * 8 + wave) * 32 + lane) << 4);
        bv.u[0] = *(const uint4*)bp;
        bv.u[1] = *(const uint4*)(bp + 8);
        acc = __builtin_amdgcn_wmma_f32_16x16x32_bf16(false, av.v, false, bv.v,
                                                      (short)0, acc, false, false);
    }
    const int n  = (wave << 4) + (lane & 15);
    const int mb = (lane >> 4) << 3;
    if (n < F_IN) {
        const float b = ob[n];
        #pragma unroll
        for (int i = 0; i < 8; ++i) {
            const int grow = row0 + mb + i;
            if (grow < rows)
                out[(size_t)grow * F_IN + n] =
                    acc[i] + b + x[((size_t)rows + grow) * F_IN + n];  // + x[:, -1]
        }
    }
}

// ---------------------------------------------------------------------------
// Host side
// ---------------------------------------------------------------------------
extern "C" void kernel_launch(void* const* d_in, const int* in_sizes, int n_in,
                              void* d_out, int out_size, void* d_ws, size_t ws_size,
                              hipStream_t stream) {
    (void)in_sizes; (void)n_in; (void)out_size; (void)ws_size;

    // d_in layout: [0]=x, [1..169]=params leaves (jax pytree order: sorted dict
    // keys, lists in order), [170..177] = latlons/edge_attrs/edge_indices.
    struct Mlp { const float *b1, *b2, *beta, *g, *w1, *w2; };
    auto M = [&](int i) {
        return Mlp{(const float*)d_in[i],     (const float*)d_in[i + 1],
                   (const float*)d_in[i + 2], (const float*)d_in[i + 3],
                   (const float*)d_in[i + 4], (const float*)d_in[i + 5]};
    };
    const float* x = (const float*)d_in[0];
    Mlp dec_emb_dst = M(1), dec_emb_edge = M(7), dec_emb_src = M(13), dec_msg = M(19);
    const float* dec_out_b = (const float*)d_in[25];
    const float* dec_out_w = (const float*)d_in[26];
    Mlp dec_upd = M(27);
    const float* e2h_tr = (const float*)d_in[33];
    Mlp enc_emb_dst = M(34), enc_emb_edge = M(40), enc_emb_src = M(46);
    Mlp enc_msg = M(52), enc_upd = M(58);
    const float* era_tr = (const float*)d_in[64];
    const float* h2e_tr = (const float*)d_in[65];
    const float* h2h_tr = (const float*)d_in[66];
    const float* h_tr   = (const float*)d_in[67];
    Mlp proc_edge = M(68);
    Mlp pmsg[8], pupd[8];
    for (int i = 0; i < 8; ++i) { pmsg[i] = M(74 + 12 * i); pupd[i] = M(80 + 12 * i); }
    const float* era_ll   = (const float*)d_in[170];
    const float* h_ll     = (const float*)d_in[171];
    const float* e2h_attr = (const float*)d_in[172];
    const float* h2e_attr = (const float*)d_in[173];
    const float* h2h_attr = (const float*)d_in[174];
    const int* e2h_idx = (const int*)d_in[175];
    const int* h2e_idx = (const int*)d_in[176];
    const int* h2h_idx = (const int*)d_in[177];

    // workspace bump allocator
    char* wp_ = (char*)d_ws;
    auto alloc = [&](size_t bytes) -> void* {
        void* p = wp_;
        wp_ += (bytes + 255) & ~(size_t)255;
        return p;
    };

    float*  XS    = (float*)alloc((size_t)N_ERA * HID * 4);  // enc era embed (xs)
    float*  XH    = (float*)alloc((size_t)N_HID * HID * 4);  // hidden state (xl)
    float*  XSD   = (float*)alloc((size_t)N_HID * HID * 4);  // dec src embed
    float*  XO    = (float*)alloc((size_t)N_ERA * HID * 4);  // dec dst embed (xo)
    float*  AGG_H = (float*)alloc((size_t)N_HID * HID * 4);
    float*  AGG_E = (float*)alloc((size_t)N_ERA * HID * 4);
    __bf16* EE2H  = (__bf16*)alloc((size_t)E_E2H * HID * 2);
    __bf16* EH2E  = (__bf16*)alloc((size_t)E_H2E * HID * 2);
    __bf16* EH2H  = (__bf16*)alloc((size_t)E_H2H * HID * 2);

    struct PW { const __bf16 *w1, *w2; };
    auto packOne = [&](const float* W, int din, int dinPad, int dout, int doutPad) {
        size_t total = (size_t)dinPad * doutPad;
        __bf16* dst = (__bf16*)alloc(total * 2);
        pack_w_kernel<<<(int)((total + 255) / 256), 256, 0, stream>>>(
            W, dst, din, dinPad, dout, doutPad);
        return (const __bf16*)dst;
    };
    auto prep = [&](const Mlp& m, int din, int dinPad) {
        return PW{packOne(m.w1, din, dinPad, HID, HID), packOne(m.w2, HID, HID, HID, HID)};
    };

    PW w_enc_src  = prep(enc_emb_src, 212, 224);
    PW w_enc_dst  = prep(enc_emb_dst, 12, 32);
    PW w_enc_edge = prep(enc_emb_edge, 11, 32);
    PW w_enc_msg  = prep(enc_msg, 384, 384);
    PW w_enc_upd  = prep(enc_upd, 256, 256);
    PW w_proc_edge = prep(proc_edge, 11, 32);
    PW w_pmsg[8], w_pupd[8];
    for (int i = 0; i < 8; ++i) {
        w_pmsg[i] = prep(pmsg[i], 384, 384);
        w_pupd[i] = prep(pupd[i], 256, 256);
    }
    PW w_dec_src  = prep(dec_emb_src, 128, 128);
    PW w_dec_dst  = prep(dec_emb_dst, 128, 128);
    PW w_dec_edge = prep(dec_emb_edge, 11, 32);
    PW w_dec_msg  = prep(dec_msg, 384, 384);
    PW w_dec_upd  = prep(dec_upd, 256, 256);
    const __bf16* w_out = packOne(dec_out_w, HID, HID, F_IN, 128);

    auto base = [&](const PW& w, const Mlp& m, int rows, int dinPad) {
        MlpArgs a{};
        a.w1p = w.w1; a.w2p = w.w2;
        a.b1 = m.b1; a.b2 = m.b2; a.g = m.g; a.beta = m.beta;
        a.rows = rows; a.dinPad = dinPad;
        return a;
    };
#define LAUNCH_MLP(GM, OM, A) \
    mlp2_kernel<GM, OM><<<((A).rows + 15) / 16, 256, 0, stream>>>(A)

    // ===== encoder =====
    { MlpArgs a = base(w_enc_src, enc_emb_src, N_ERA, 224);
      a.srcA = x; a.srcB = era_ll; a.srcC = era_tr; a.outF = XS;
      LAUNCH_MLP(G_ERA, O_F32, a); }
    { MlpArgs a = base(w_enc_dst, enc_emb_dst, N_HID, 32);
      a.srcA = h_ll; a.srcB = h_tr; a.outF = XH;
      LAUNCH_MLP(G_NODEF, O_F32, a); }
    { MlpArgs a = base(w_enc_edge, enc_emb_edge, E_E2H, 32);
      a.srcA = e2h_attr; a.srcB = e2h_tr; a.outB = EE2H;
      LAUNCH_MLP(G_EDGEF, O_BF16, a); }
    zero_kernel<<<(N_HID * HID + 255) / 256, 256, 0, stream>>>(AGG_H, N_HID * HID);
    { MlpArgs a = base(w_enc_msg, enc_msg, E_E2H, 384);
      a.srcA = XS; a.srcB = XH; a.srcE = EE2H;
      a.sidx = e2h_idx; a.didx = e2h_idx + E_E2H; a.outF = AGG_H;
      LAUNCH_MLP(G_CAT3, O_SCATTER, a); }
    { MlpArgs a = base(w_enc_upd, enc_upd, N_HID, 256);
      a.srcA = XH; a.srcB = AGG_H; a.resBase = XH; a.outF = XH;
      LAUNCH_MLP(G_CAT2, O_RESID, a); }

    // ===== processor =====
    { MlpArgs a = base(w_proc_edge, proc_edge, E_H2H, 32);
      a.srcA = h2h_attr; a.srcB = h2h_tr; a.outB = EH2H;
      LAUNCH_MLP(G_EDGEF, O_BF16, a); }
    for (int L = 0; L < 8; ++L) {
        zero_kernel<<<(N_HID * HID + 255) / 256, 256, 0, stream>>>(AGG_H, N_HID * HID);
        { MlpArgs a = base(w_pmsg[L], pmsg[L], E_H2H, 384);
          a.srcA = XH; a.srcB = XH; a.srcE = EH2H;
          a.sidx = h2h_idx; a.didx = h2h_idx + E_H2H; a.outF = AGG_H;
          LAUNCH_MLP(G_CAT3, O_SCATTER, a); }
        { MlpArgs a = base(w_pupd[L], pupd[L], N_HID, 256);
          a.srcA = XH; a.srcB = AGG_H; a.resBase = XH; a.outF = XH;
          LAUNCH_MLP(G_CAT2, O_RESID, a); }
    }

    // ===== decoder =====
    { MlpArgs a = base(w_dec_src, dec_emb_src, N_HID, 128);
      a.srcA = XH; a.outF = XSD;
      LAUNCH_MLP(G_PLAIN, O_F32, a); }
    { MlpArgs a = base(w_dec_dst, dec_emb_dst, N_ERA, 128);
      a.srcA = XS; a.outF = XO;
      LAUNCH_MLP(G_PLAIN, O_F32, a); }
    { MlpArgs a = base(w_dec_edge, dec_emb_edge, E_H2E, 32);
      a.srcA = h2e_attr; a.srcB = h2e_tr; a.outB = EH2E;
      LAUNCH_MLP(G_EDGEF, O_BF16, a); }
    zero_kernel<<<(N_ERA * HID + 255) / 256, 256, 0, stream>>>(AGG_E, N_ERA * HID);
    { MlpArgs a = base(w_dec_msg, dec_msg, E_H2E, 384);
      a.srcA = XSD; a.srcB = XO; a.srcE = EH2E;
      a.sidx = h2e_idx; a.didx = h2e_idx + E_H2E; a.outF = AGG_E;
      LAUNCH_MLP(G_CAT3, O_SCATTER, a); }
    { MlpArgs a = base(w_dec_upd, dec_upd, N_ERA, 256);
      a.srcA = XO; a.srcB = AGG_E; a.resBase = XO; a.outF = XO;
      LAUNCH_MLP(G_CAT2, O_RESID, a); }

    // ===== output head + residual =====
    out_gemm_kernel<<<(N_ERA + 15) / 16, 256, 0, stream>>>(
        XO, w_out, dec_out_b, x, (float*)d_out, N_ERA);
#undef LAUNCH_MLP
}
